// CA_Module_38912403702093
// MI455X (gfx1250) — compile-verified, matrix operand order checked by reading, not run on hardware
//
#include <hip/hip_runtime.h>
#include <hip/hip_bf16.h>

#define BATCH 16
#define CCH   256
#define NPIX  4096   // 64*64

typedef __attribute__((ext_vector_type(16))) __bf16 v16bf;
typedef __attribute__((ext_vector_type(8)))  __bf16 v8bf;
typedef __attribute__((ext_vector_type(8)))  float  v8f;
typedef __attribute__((ext_vector_type(2)))  float  v2f;

union Frag16 { v16bf v; v8bf h[2]; };

// ---------------------------------------------------------------------------
// K0: fused x(f32) -> Xbf(bf16 row-major) + XbfT(bf16 transposed) + partial
// row sums. One 32x32 tile per block; x is read exactly once. Each 32-thread
// row of the block is one wave32, so row sums use in-wave shfl reductions
// (deterministic, no float atomics).
// ---------------------------------------------------------------------------
__global__ void __launch_bounds__(256) cvt_fused_kernel(const float* __restrict__ x,
                                                        __bf16* __restrict__ Xbf,
                                                        __bf16* __restrict__ XbfT,
                                                        float* __restrict__ spart) {
  __shared__ __bf16 tile[32][33];        // +1 pad to dodge bank conflicts
  const int id = blockIdx.x;             // b*1024 + ct*128 + nt
  const int b  = id >> 10;
  const int t  = id & 1023;
  const int ct = t >> 7;                 // channel tile 0..7
  const int nt = t & 127;                // pixel tile   0..127
  const int tx = threadIdx.x & 31;
  const int ty = threadIdx.x >> 5;       // 0..7 (one wave per ty)
  const float* xb = x + (size_t)b * CCH * NPIX;
  __bf16* Xb = Xbf + (size_t)b * CCH * NPIX;

  float sums[4];
  for (int i = 0; i < 4; ++i) {
    int c = ct * 32 + ty + 8 * i;
    float v = xb[(size_t)c * NPIX + nt * 32 + tx];
    __bf16 h = (__bf16)v;
    Xb[(size_t)c * NPIX + nt * 32 + tx] = h;
    tile[ty + 8 * i][tx] = h;
    sums[i] = v;
  }
  // wave32 reduction across tx for each of the 4 rows this wave owns
  for (int i = 0; i < 4; ++i)
    for (int m = 16; m >= 1; m >>= 1)
      sums[i] += __shfl_xor(sums[i], m, 32);
  if (tx == 0) {
    for (int i = 0; i < 4; ++i) {
      int c = ct * 32 + ty + 8 * i;
      spart[((size_t)b * CCH + c) * 128 + nt] = sums[i];
    }
  }
  __syncthreads();
  __bf16* Tb = XbfT + (size_t)b * NPIX * CCH;
  for (int i = 0; i < 4; ++i) {
    int n = nt * 32 + ty + 8 * i;
    Tb[(size_t)n * CCH + ct * 32 + tx] = tile[tx][ty + 8 * i];
  }
}

// K0b: svec[row] = sum over 128 tile partials
__global__ void __launch_bounds__(128) rowsum_reduce_kernel(const float* __restrict__ spart,
                                                            float* __restrict__ svec) {
  __shared__ float red[128];
  const int row = blockIdx.x;            // b*C + c
  const int tid = threadIdx.x;
  red[tid] = spart[(size_t)row * 128 + tid];
  __syncthreads();
  for (int s = 64; s > 0; s >>= 1) {
    if (tid < s) red[tid] += red[tid + s];
    __syncthreads();
  }
  if (tid == 0) svec[row] = red[0];
}

// ---------------------------------------------------------------------------
// K1: Gram matrix G[b] = X[b] * X[b]^T, [256x256] f32, K = 4096, bf16 WMMA.
// One wave per 32x32 output tile, 2x2 register blocking.
// ---------------------------------------------------------------------------
__global__ void __launch_bounds__(32) gram_kernel(const __bf16* __restrict__ Xbf,
                                                  float* __restrict__ G) {
  const int id = blockIdx.x;             // b*64 + tile
  const int b  = id >> 6;
  const int t  = id & 63;
  const int tm = (t >> 3) << 5;
  const int tn = (t & 7) << 5;
  const __bf16* Xb = Xbf + (size_t)b * CCH * NPIX;
  const int lane = threadIdx.x;
  const int r = lane & 15;
  const int g = lane >> 4;

  v8f acc00 = {}, acc01 = {}, acc10 = {}, acc11 = {};
  const __bf16* rowA0 = Xb + (size_t)(tm + r) * NPIX;
  const __bf16* rowA1 = Xb + (size_t)(tm + 16 + r) * NPIX;
  const __bf16* rowB0 = Xb + (size_t)(tn + r) * NPIX;
  const __bf16* rowB1 = Xb + (size_t)(tn + 16 + r) * NPIX;

  for (int k0 = 0; k0 < NPIX; k0 += 32) {
    const int ko = k0 + 8 * g;
    Frag16 fa0, fa1, fb0, fb1;
    fa0.h[0] = *(const v8bf*)(rowA0 + ko);      fa0.h[1] = *(const v8bf*)(rowA0 + ko + 16);
    fa1.h[0] = *(const v8bf*)(rowA1 + ko);      fa1.h[1] = *(const v8bf*)(rowA1 + ko + 16);
    fb0.h[0] = *(const v8bf*)(rowB0 + ko);      fb0.h[1] = *(const v8bf*)(rowB0 + ko + 16);
    fb1.h[0] = *(const v8bf*)(rowB1 + ko);      fb1.h[1] = *(const v8bf*)(rowB1 + ko + 16);
    __builtin_prefetch(rowA0 + ko + 64, 0, 1);
    __builtin_prefetch(rowB0 + ko + 64, 0, 1);
    acc00 = __builtin_amdgcn_wmma_f32_16x16x32_bf16(false, fa0.v, false, fb0.v, (short)0, acc00, false, false);
    acc01 = __builtin_amdgcn_wmma_f32_16x16x32_bf16(false, fa0.v, false, fb1.v, (short)0, acc01, false, false);
    acc10 = __builtin_amdgcn_wmma_f32_16x16x32_bf16(false, fa1.v, false, fb0.v, (short)0, acc10, false, false);
    acc11 = __builtin_amdgcn_wmma_f32_16x16x32_bf16(false, fa1.v, false, fb1.v, (short)0, acc11, false, false);
  }

  float* Gb = G + (size_t)b * CCH * CCH;
  const int n0 = tn + r;
  for (int j = 0; j < 8; ++j) {
    int m0 = tm + j + 8 * g;
    int m1 = m0 + 16;
    Gb[(size_t)m0 * CCH + n0]      = acc00[j];
    Gb[(size_t)m0 * CCH + n0 + 16] = acc01[j];
    Gb[(size_t)m1 * CCH + n0]      = acc10[j];
    Gb[(size_t)m1 * CCH + n0 + 16] = acc11[j];
  }
}

// ---------------------------------------------------------------------------
// K2v: u = Wq*s, w = Wk*s per batch
// ---------------------------------------------------------------------------
__global__ void __launch_bounds__(256) matvec_uw_kernel(const float* __restrict__ Wq,
                                                        const float* __restrict__ Wk,
                                                        const float* __restrict__ svec,
                                                        float* __restrict__ uvec,
                                                        float* __restrict__ wvec) {
  const int b = blockIdx.x;
  const int o = threadIdx.x;
  const float* sb = svec + (size_t)b * CCH;
  float su = 0.f, sw = 0.f;
  for (int i = 0; i < CCH; ++i) {
    float si = sb[i];
    su += Wq[(size_t)o * CCH + i] * si;
    sw += Wk[(size_t)o * CCH + i] * si;
  }
  uvec[(size_t)b * CCH + o] = su;
  wvec[(size_t)b * CCH + o] = sw;
}

// ---------------------------------------------------------------------------
// f32 WMMA 16x16x4 GEMMs (all M=N=K=256), one wave per 32x32 tile
// ---------------------------------------------------------------------------
__global__ void __launch_bounds__(32) gemm_TG_kernel(const float* __restrict__ Wq,
                                                     const float* __restrict__ G,
                                                     float* __restrict__ T) {
  const int id = blockIdx.x;
  const int b  = id >> 6;
  const int t  = id & 63;
  const int tm = (t >> 3) << 5;
  const int tn = (t & 7) << 5;
  const float* Bb = G + (size_t)b * CCH * CCH;
  float* Db = T + (size_t)b * CCH * CCH;
  const int lane = threadIdx.x, r = lane & 15, g = lane >> 4;

  v8f acc00 = {}, acc01 = {}, acc10 = {}, acc11 = {};
  for (int k0 = 0; k0 < CCH; k0 += 4) {
    const int k = k0 + 2 * g;
    v2f fa0 = *(const v2f*)(Wq + (size_t)(tm + r) * CCH + k);
    v2f fa1 = *(const v2f*)(Wq + (size_t)(tm + 16 + r) * CCH + k);
    v2f fb0, fb1;
    fb0.x = Bb[(size_t)k * CCH + tn + r];
    fb0.y = Bb[(size_t)(k + 1) * CCH + tn + r];
    fb1.x = Bb[(size_t)k * CCH + tn + 16 + r];
    fb1.y = Bb[(size_t)(k + 1) * CCH + tn + 16 + r];
    acc00 = __builtin_amdgcn_wmma_f32_16x16x4_f32(false, fa0, false, fb0, (short)0, acc00, false, false);
    acc01 = __builtin_amdgcn_wmma_f32_16x16x4_f32(false, fa0, false, fb1, (short)0, acc01, false, false);
    acc10 = __builtin_amdgcn_wmma_f32_16x16x4_f32(false, fa1, false, fb0, (short)0, acc10, false, false);
    acc11 = __builtin_amdgcn_wmma_f32_16x16x4_f32(false, fa1, false, fb1, (short)0, acc11, false, false);
  }
  const int n0 = tn + r;
  for (int j = 0; j < 8; ++j) {
    int m0 = tm + j + 8 * g, m1 = m0 + 16;
    Db[(size_t)m0 * CCH + n0]      = acc00[j];
    Db[(size_t)m0 * CCH + n0 + 16] = acc01[j];
    Db[(size_t)m1 * CCH + n0]      = acc10[j];
    Db[(size_t)m1 * CCH + n0 + 16] = acc11[j];
  }
}

__global__ void __launch_bounds__(32) gemm_E_kernel(const float* __restrict__ T,
                                                    const float* __restrict__ Wk,
                                                    const float* __restrict__ uvec,
                                                    const float* __restrict__ wvec,
                                                    const float* __restrict__ bq,
                                                    const float* __restrict__ bk,
                                                    float* __restrict__ E) {
  const int id = blockIdx.x;
  const int b  = id >> 6;
  const int t  = id & 63;
  const int tm = (t >> 3) << 5;
  const int tn = (t & 7) << 5;
  const float* Ab = T + (size_t)b * CCH * CCH;
  float* Db = E + (size_t)b * CCH * CCH;
  const float* ub = uvec + (size_t)b * CCH;
  const float* wb = wvec + (size_t)b * CCH;
  const int lane = threadIdx.x, r = lane & 15, g = lane >> 4;

  v8f acc00 = {}, acc01 = {}, acc10 = {}, acc11 = {};
  for (int k0 = 0; k0 < CCH; k0 += 4) {
    const int k = k0 + 2 * g;
    v2f fa0 = *(const v2f*)(Ab + (size_t)(tm + r) * CCH + k);
    v2f fa1 = *(const v2f*)(Ab + (size_t)(tm + 16 + r) * CCH + k);
    v2f fb0 = *(const v2f*)(Wk + (size_t)(tn + r) * CCH + k);        // B = Wk^T
    v2f fb1 = *(const v2f*)(Wk + (size_t)(tn + 16 + r) * CCH + k);
    acc00 = __builtin_amdgcn_wmma_f32_16x16x4_f32(false, fa0, false, fb0, (short)0, acc00, false, false);
    acc01 = __builtin_amdgcn_wmma_f32_16x16x4_f32(false, fa0, false, fb1, (short)0, acc01, false, false);
    acc10 = __builtin_amdgcn_wmma_f32_16x16x4_f32(false, fa1, false, fb0, (short)0, acc10, false, false);
    acc11 = __builtin_amdgcn_wmma_f32_16x16x4_f32(false, fa1, false, fb1, (short)0, acc11, false, false);
  }
  const int n0 = tn + r, n1 = n0 + 16;
  const float bk0 = bk[n0], bk1 = bk[n1];
  const float w0 = wb[n0], w1 = wb[n1];
  for (int j = 0; j < 8; ++j) {
    int m0 = tm + j + 8 * g, m1 = m0 + 16;
    float u0 = ub[m0], u1 = ub[m1];
    float q0 = bq[m0], q1 = bq[m1];
    Db[(size_t)m0 * CCH + n0] = acc00[j] + u0 * bk0 + q0 * w0 + (float)NPIX * q0 * bk0;
    Db[(size_t)m0 * CCH + n1] = acc01[j] + u0 * bk1 + q0 * w1 + (float)NPIX * q0 * bk1;
    Db[(size_t)m1 * CCH + n0] = acc10[j] + u1 * bk0 + q1 * w0 + (float)NPIX * q1 * bk0;
    Db[(size_t)m1 * CCH + n1] = acc11[j] + u1 * bk1 + q1 * w1 + (float)NPIX * q1 * bk1;
  }
}

__global__ void __launch_bounds__(32) gemm_AV_kernel(const float* __restrict__ A,
                                                     const float* __restrict__ Wv,
                                                     __bf16* __restrict__ Mbf) {
  const int id = blockIdx.x;
  const int b  = id >> 6;
  const int t  = id & 63;
  const int tm = (t >> 3) << 5;
  const int tn = (t & 7) << 5;
  const float* Ab = A + (size_t)b * CCH * CCH;
  __bf16* Db = Mbf + (size_t)b * CCH * CCH;
  const int lane = threadIdx.x, r = lane & 15, g = lane >> 4;

  v8f acc00 = {}, acc01 = {}, acc10 = {}, acc11 = {};
  for (int k0 = 0; k0 < CCH; k0 += 4) {
    const int k = k0 + 2 * g;
    v2f fa0 = *(const v2f*)(Ab + (size_t)(tm + r) * CCH + k);
    v2f fa1 = *(const v2f*)(Ab + (size_t)(tm + 16 + r) * CCH + k);
    v2f fb0, fb1;
    fb0.x = Wv[(size_t)k * CCH + tn + r];
    fb0.y = Wv[(size_t)(k + 1) * CCH + tn + r];
    fb1.x = Wv[(size_t)k * CCH + tn + 16 + r];
    fb1.y = Wv[(size_t)(k + 1) * CCH + tn + 16 + r];
    acc00 = __builtin_amdgcn_wmma_f32_16x16x4_f32(false, fa0, false, fb0, (short)0, acc00, false, false);
    acc01 = __builtin_amdgcn_wmma_f32_16x16x4_f32(false, fa0, false, fb1, (short)0, acc01, false, false);
    acc10 = __builtin_amdgcn_wmma_f32_16x16x4_f32(false, fa1, false, fb0, (short)0, acc10, false, false);
    acc11 = __builtin_amdgcn_wmma_f32_16x16x4_f32(false, fa1, false, fb1, (short)0, acc11, false, false);
  }
  const int n0 = tn + r;
  for (int j = 0; j < 8; ++j) {
    int m0 = tm + j + 8 * g, m1 = m0 + 16;
    Db[(size_t)m0 * CCH + n0]      = (__bf16)acc00[j];
    Db[(size_t)m0 * CCH + n0 + 16] = (__bf16)acc01[j];
    Db[(size_t)m1 * CCH + n0]      = (__bf16)acc10[j];
    Db[(size_t)m1 * CCH + n0 + 16] = (__bf16)acc11[j];
  }
}

// ---------------------------------------------------------------------------
// K3: row softmax over E[b] rows of 256 (in place)
// ---------------------------------------------------------------------------
__global__ void __launch_bounds__(256) softmax_kernel(float* __restrict__ E) {
  __shared__ float red[256];
  const int row = blockIdx.x;            // b*C + i
  const int tid = threadIdx.x;
  float* e = E + (size_t)row * CCH;
  float v = e[tid];
  red[tid] = v;
  __syncthreads();
  for (int s = 128; s > 0; s >>= 1) {
    if (tid < s) red[tid] = fmaxf(red[tid], red[tid + s]);
    __syncthreads();
  }
  const float m = red[0];
  __syncthreads();
  float ex = __expf(v - m);
  red[tid] = ex;
  __syncthreads();
  for (int s = 128; s > 0; s >>= 1) {
    if (tid < s) red[tid] += red[tid + s];
    __syncthreads();
  }
  e[tid] = ex / red[0];
}

// ---------------------------------------------------------------------------
// K3b: c[b][i] = sum_j attn[b][i][j] * bv[j]
// ---------------------------------------------------------------------------
__global__ void __launch_bounds__(256) attn_bv_kernel(const float* __restrict__ A,
                                                      const float* __restrict__ bv,
                                                      float* __restrict__ cvec) {
  const int b = blockIdx.x;
  const int i = threadIdx.x;
  const float* a = A + ((size_t)b * CCH + i) * CCH;
  float sum = 0.f;
  for (int j = 0; j < CCH; ++j) sum += a[j] * bv[j];
  cvec[(size_t)b * CCH + i] = sum;
}

// ---------------------------------------------------------------------------
// K4: y = gamma*(Mbf[b]*X[b] + c*1^T) + x ; bf16 WMMA, K=256, B = XbfT rows
// ---------------------------------------------------------------------------
__global__ void __launch_bounds__(32) out_gemm_kernel(const __bf16* __restrict__ Mbf,
                                                      const __bf16* __restrict__ XbfT,
                                                      const float* __restrict__ x,
                                                      const float* __restrict__ cvec,
                                                      const float* __restrict__ gamma,
                                                      float* __restrict__ y) {
  const int id = blockIdx.x;             // b*(8*128) + mt*128 + nt
  const int b  = id >> 10;
  const int t  = id & 1023;
  const int tm = (t >> 7) << 5;          // 0..224
  const int tn = (t & 127) << 5;         // 0..4064
  const __bf16* Ab = Mbf + (size_t)b * CCH * CCH;
  const __bf16* Bb = XbfT + (size_t)b * NPIX * CCH;
  const int lane = threadIdx.x, r = lane & 15, g = lane >> 4;

  v8f acc00 = {}, acc01 = {}, acc10 = {}, acc11 = {};
  const __bf16* rowA0 = Ab + (size_t)(tm + r) * CCH;
  const __bf16* rowA1 = Ab + (size_t)(tm + 16 + r) * CCH;
  const __bf16* rowB0 = Bb + (size_t)(tn + r) * CCH;
  const __bf16* rowB1 = Bb + (size_t)(tn + 16 + r) * CCH;

  for (int k0 = 0; k0 < CCH; k0 += 32) {
    const int ko = k0 + 8 * g;
    Frag16 fa0, fa1, fb0, fb1;
    fa0.h[0] = *(const v8bf*)(rowA0 + ko);      fa0.h[1] = *(const v8bf*)(rowA0 + ko + 16);
    fa1.h[0] = *(const v8bf*)(rowA1 + ko);      fa1.h[1] = *(const v8bf*)(rowA1 + ko + 16);
    fb0.h[0] = *(const v8bf*)(rowB0 + ko);      fb0.h[1] = *(const v8bf*)(rowB0 + ko + 16);
    fb1.h[0] = *(const v8bf*)(rowB1 + ko);      fb1.h[1] = *(const v8bf*)(rowB1 + ko + 16);
    __builtin_prefetch(rowB0 + ko + 64, 0, 1);
    acc00 = __builtin_amdgcn_wmma_f32_16x16x32_bf16(false, fa0.v, false, fb0.v, (short)0, acc00, false, false);
    acc01 = __builtin_amdgcn_wmma_f32_16x16x32_bf16(false, fa0.v, false, fb1.v, (short)0, acc01, false, false);
    acc10 = __builtin_amdgcn_wmma_f32_16x16x32_bf16(false, fa1.v, false, fb0.v, (short)0, acc10, false, false);
    acc11 = __builtin_amdgcn_wmma_f32_16x16x32_bf16(false, fa1.v, false, fb1.v, (short)0, acc11, false, false);
  }

  const float gam = *gamma;
  const float* cb = cvec + (size_t)b * CCH;
  const int n0 = tn + r, n1 = n0 + 16;
  for (int j = 0; j < 8; ++j) {
    int m0 = tm + j + 8 * g, m1 = m0 + 16;
    size_t r0 = ((size_t)b * CCH + m0) * NPIX;
    size_t r1 = ((size_t)b * CCH + m1) * NPIX;
    float c0 = cb[m0], c1 = cb[m1];
    y[r0 + n0] = gam * (acc00[j] + c0) + x[r0 + n0];
    y[r0 + n1] = gam * (acc01[j] + c0) + x[r0 + n1];
    y[r1 + n0] = gam * (acc10[j] + c1) + x[r1 + n0];
    y[r1 + n1] = gam * (acc11[j] + c1) + x[r1 + n1];
  }
}

// ---------------------------------------------------------------------------
extern "C" void kernel_launch(void* const* d_in, const int* in_sizes, int n_in,
                              void* d_out, int out_size, void* d_ws, size_t ws_size,
                              hipStream_t stream) {
  const float* x     = (const float*)d_in[0];
  const float* wq    = (const float*)d_in[1];
  const float* bq    = (const float*)d_in[2];
  const float* wk    = (const float*)d_in[3];
  const float* bk    = (const float*)d_in[4];
  const float* wv    = (const float*)d_in[5];
  const float* bv    = (const float*)d_in[6];
  const float* gamma = (const float*)d_in[7];
  float* out = (float*)d_out;

  char* ws = (char*)d_ws;
  size_t off = 0;
  __bf16* Xbf  = (__bf16*)(ws + off); off += (size_t)BATCH * CCH * NPIX * 2;  // 32 MiB
  __bf16* XbfT = (__bf16*)(ws + off); off += (size_t)BATCH * CCH * NPIX * 2;  // 32 MiB
  float* spart = (float*)(ws + off);  off += (size_t)BATCH * CCH * 128 * 4;   // 2 MiB
  float* svec  = (float*)(ws + off);  off += (size_t)BATCH * CCH * 4;
  float* uvec  = (float*)(ws + off);  off += (size_t)BATCH * CCH * 4;
  float* wvec  = (float*)(ws + off);  off += (size_t)BATCH * CCH * 4;
  float* cvec  = (float*)(ws + off);  off += (size_t)BATCH * CCH * 4;
  float* Gbuf  = (float*)(ws + off);  off += (size_t)BATCH * CCH * CCH * 4;   // G, later E/attn
  float* Tbuf  = (float*)(ws + off);  off += (size_t)BATCH * CCH * CCH * 4;   // T, later Mbf
  float* Ebuf  = Gbuf;                 // E overwrites G after T is computed
  __bf16* Mbf  = (__bf16*)Tbuf;        // Mbf overwrites T after E is computed
  (void)ws_size; (void)in_sizes; (void)n_in; (void)out_size;

  // 1. fused bf16 convert + transpose + partial row sums (x read once)
  cvt_fused_kernel<<<BATCH * 8 * 128, 256, 0, stream>>>(x, Xbf, XbfT, spart);
  rowsum_reduce_kernel<<<BATCH * CCH, 128, 0, stream>>>(spart, svec);
  // 2. Gram matrices G = X X^T  (bf16 WMMA, K=4096)
  gram_kernel<<<BATCH * 64, 32, 0, stream>>>(Xbf, Gbuf);
  // 3. u = Wq s, w = Wk s
  matvec_uw_kernel<<<BATCH, 256, 0, stream>>>(wq, wk, svec, uvec, wvec);
  // 4. T = Wq G  (f32 WMMA)
  gemm_TG_kernel<<<BATCH * 64, 32, 0, stream>>>(wq, Gbuf, Tbuf);
  // 5. E = T Wk^T + u bk^T + bq w^T + N bq bk^T  (f32 WMMA)
  gemm_E_kernel<<<BATCH * 64, 32, 0, stream>>>(Tbuf, wk, uvec, wvec, bq, bk, Ebuf);
  // 6. softmax rows (in place -> attn)
  softmax_kernel<<<BATCH * CCH, 256, 0, stream>>>(Ebuf);
  // 7. c = attn bv
  attn_bv_kernel<<<BATCH, 256, 0, stream>>>(Ebuf, bv, cvec);
  // 8. M = attn Wv -> bf16  (f32 WMMA)
  gemm_AV_kernel<<<BATCH * 64, 32, 0, stream>>>(Ebuf, wv, Mbf);
  // 9. y = gamma*(M X + c 1^T) + x  (bf16 WMMA, K=256)
  out_gemm_kernel<<<BATCH * 8 * 128, 32, 0, stream>>>(Mbf, XbfT, x, cvec, gamma, out);
}
